// MoELayer_1906965480199
// MI455X (gfx1250) — matrix operand order repaired and checked by reading, not compiled, not run on hardware
//
#include <hip/hip_runtime.h>
#include <hip/hip_bf16.h>

// ---------------------------------------------------------------------------
// MoE dense-expert layer for MI455X (gfx1250, wave32, WMMA bf16 path)
// B=4, T=2048, C=1024, D=2048, E=8 ; N = B*T = 8192 tokens
// Weights are pre-transposed (N-major, K-contiguous) during the one-time
// fp32->bf16 convert, so both GEMM operands stage into LDS with plain b128
// copies. Double-buffered, branch-free steady-state WMMA mainloop.
// ---------------------------------------------------------------------------

#define NTOK 8192
#define CDIM 1024
#define DDIM 2048
#define NEXP 8

typedef __attribute__((ext_vector_type(16))) __bf16 v16bf;
typedef __attribute__((ext_vector_type(8)))  float  v8f;

struct U4Pair { uint4 lo, hi; };

static __device__ __forceinline__ v16bf make_frag(const uint4& lo, const uint4& hi) {
    U4Pair p{lo, hi};
    return __builtin_bit_cast(v16bf, p);
}

static __device__ __forceinline__ unsigned short f2bf(float f) {
    unsigned int u = __float_as_uint(f);
    // round-to-nearest-even to bf16
    unsigned int r = (u + 0x7FFFu + ((u >> 16) & 1u)) >> 16;
    return (unsigned short)r;
}

// ---------------------------------------------------------------------------
// fp32 -> bf16 conversion, vectorized x4 (for x, which stays K-contiguous)
// ---------------------------------------------------------------------------
__global__ __launch_bounds__(256) void cvt_f32_bf16(const float* __restrict__ in,
                                                    unsigned short* __restrict__ out,
                                                    int n4) {
    int i = blockIdx.x * blockDim.x + threadIdx.x;
    if (i < n4) {
        float4 f = ((const float4*)in)[i];
        union { unsigned short s[4]; uint2 u; } p;
        p.s[0] = f2bf(f.x); p.s[1] = f2bf(f.y);
        p.s[2] = f2bf(f.z); p.s[3] = f2bf(f.w);
        ((uint2*)out)[i] = p.u;
    }
}

// ---------------------------------------------------------------------------
// fp32 [E][K][N] -> bf16 [E][N][K] transposing convert, 64x64 tiles via LDS.
// Coalesced float4 reads, coalesced b128 writes.
// ---------------------------------------------------------------------------
__global__ __launch_bounds__(256) void cvt_transpose_bf16(
    const float* __restrict__ in,          // [E][K][N] f32
    unsigned short* __restrict__ out,      // [E][N][K] bf16
    int K, int N) {
    const int e  = blockIdx.z;
    const float*          src = in  + (size_t)e * K * N;
    unsigned short*       dst = out + (size_t)e * K * N;
    const int kt = blockIdx.y * 64;
    const int nt = blockIdx.x * 64;
    __shared__ __align__(8) unsigned short lds[64][68];   // pad to 68 elems
    const int tid = threadIdx.x;

#pragma unroll
    for (int i = 0; i < 4; ++i) {
        int idx = tid + i * 256;                   // 1024 float4 units
        int r   = idx >> 4;                        // k-within-tile 0..63
        int cq  = idx & 15;                        // float4 quad 0..15
        float4 f = *(const float4*)&src[(size_t)(kt + r) * N + nt + cq * 4];
        union { unsigned short s[4]; uint2 u; } p;
        p.s[0] = f2bf(f.x); p.s[1] = f2bf(f.y);
        p.s[2] = f2bf(f.z); p.s[3] = f2bf(f.w);
        *(uint2*)&lds[r][cq * 4] = p.u;
    }
    __syncthreads();
#pragma unroll
    for (int i = 0; i < 2; ++i) {
        int u  = tid + i * 256;                    // 512 b128 units
        int c  = u >> 3;                           // n-within-tile 0..63
        int ch = u & 7;                            // k chunk of 8
        union { unsigned short s[8]; uint4 v; } p;
#pragma unroll
        for (int j = 0; j < 8; ++j) p.s[j] = lds[ch * 8 + j][c];
        *(uint4*)&dst[(size_t)(nt + c) * K + kt + ch * 8] = p.v;
    }
}

// ---------------------------------------------------------------------------
// Router: probs[token][e] = softmax_e( x . w_router + b_router )
// 8 lanes per token (E=8); softmax across the 8-lane group via shfl_xor.
// ---------------------------------------------------------------------------
__global__ __launch_bounds__(256) void router_kernel(const float* __restrict__ x,
                                                     const float* __restrict__ wr,
                                                     const float* __restrict__ br,
                                                     float* __restrict__ probs) {
    int tid   = threadIdx.x;
    int token = blockIdx.x * 32 + (tid >> 3);
    int e     = tid & 7;
    const float* xr = x + (size_t)token * CDIM;
    float acc = 0.f;
#pragma unroll 8
    for (int c = 0; c < CDIM; ++c)
        acc += xr[c] * wr[c * NEXP + e];
    acc += br[e];
    float m = acc;
    m = fmaxf(m, __shfl_xor(m, 1, 8));
    m = fmaxf(m, __shfl_xor(m, 2, 8));
    m = fmaxf(m, __shfl_xor(m, 4, 8));
    float ex = __expf(acc - m);
    float s = ex;
    s += __shfl_xor(s, 1, 8);
    s += __shfl_xor(s, 2, 8);
    s += __shfl_xor(s, 4, 8);
    probs[token * NEXP + e] = ex / s;
}

// ---------------------------------------------------------------------------
// WMMA GEMM:  Out = epilogue( A[M=8192, KD] * Bt^T + bias ),  Bt = [ND, KD]
//   EPI==0 : Hout(bf16) = gelu_exact(.)                       (GEMM1)
//   EPI==1 : Fout(f32) (+)= probs[row][expert] * (.)          (GEMM2)
// Block tile 128x128, BK=32 (one WMMA K-step per LDS stage), double-buffered,
// branch-free steady state with peeled last iteration.
// 8 waves per block in a 4(M) x 2(N) grid; each wave owns 2x4 16x16 tiles.
// ---------------------------------------------------------------------------
template <int KD, int ND, int EPI>
__global__ __launch_bounds__(256) void gemm_wmma(
    const unsigned short* __restrict__ A,    // [8192, KD] bf16 (K-contiguous)
    const unsigned short* __restrict__ Bt,   // [ND, KD]   bf16 (K-contiguous)
    const float* __restrict__ bias,          // [ND]
    unsigned short* __restrict__ Hout,       // bf16 out (EPI==0)
    float* __restrict__ Fout,                // f32 out  (EPI==1)
    const float* __restrict__ probs,         // [8192, 8] (EPI==1)
    int expert, int initOut) {

    const int tid  = threadIdx.x;
    const int wave = tid >> 5;
    const int lane = tid & 31;
    const int wm   = wave & 3;       // 4 waves along M (32 rows each)
    const int wn   = wave >> 2;      // 2 waves along N (64 cols each)
    const int lr   = lane & 15;      // lane row/col inside 16
    const int lh   = lane >> 4;      // lane half-select
    const int m0   = blockIdx.y * 128;
    const int n0   = blockIdx.x * 128;

    // padded LDS: row stride 40 elements (80B) to spread banks; 2 buffers
    __shared__ __align__(16) unsigned short ldsA[2][128 * 40]; // [row][k]
    __shared__ __align__(16) unsigned short ldsB[2][128 * 40]; // [col][k]

    // ---- per-thread staging coordinates (identical pattern for A and B) ----
    const int r0 = tid >> 2,          h0 = tid & 3;
    const int r1 = (tid + 256) >> 2,  h1 = (tid + 256) & 3;

    v8f acc[2][4];
#pragma unroll
    for (int mm = 0; mm < 2; ++mm)
#pragma unroll
        for (int nn = 0; nn < 4; ++nn)
            acc[mm][nn] = (v8f){0.f, 0.f, 0.f, 0.f, 0.f, 0.f, 0.f, 0.f};

    auto loadGlobal = [&](int k0, uint4& a0, uint4& a1, uint4& b0, uint4& b1) {
        a0 = *(const uint4*)&A [(size_t)(m0 + r0) * KD + k0 + h0 * 8];
        a1 = *(const uint4*)&A [(size_t)(m0 + r1) * KD + k0 + h1 * 8];
        b0 = *(const uint4*)&Bt[(size_t)(n0 + r0) * KD + k0 + h0 * 8];
        b1 = *(const uint4*)&Bt[(size_t)(n0 + r1) * KD + k0 + h1 * 8];
    };
    auto storeLds = [&](int buf, const uint4& a0, const uint4& a1,
                        const uint4& b0, const uint4& b1) {
        *(uint4*)&ldsA[buf][r0 * 40 + h0 * 8] = a0;
        *(uint4*)&ldsA[buf][r1 * 40 + h1 * 8] = a1;
        *(uint4*)&ldsB[buf][r0 * 40 + h0 * 8] = b0;
        *(uint4*)&ldsB[buf][r1 * 40 + h1 * 8] = b1;
    };
    auto compute = [&](int buf) {
        v16bf fa[2], fb[4];
#pragma unroll
        for (int mm = 0; mm < 2; ++mm) {
            int row  = wm * 32 + mm * 16 + lr;
            int base = row * 40 + lh * 8;                     // K 0-7 | 8-15
            fa[mm] = make_frag(*(const uint4*)&ldsA[buf][base],
                               *(const uint4*)&ldsA[buf][base + 16]); // K 16-23|24-31
        }
#pragma unroll
        for (int nn = 0; nn < 4; ++nn) {
            int col  = wn * 64 + nn * 16 + lr;
            int base = col * 40 + lh * 8;                     // K 0-7 | 8-15
            fb[nn] = make_frag(*(const uint4*)&ldsB[buf][base],
                               *(const uint4*)&ldsB[buf][base + 16]); // K 16-23|24-31
        }
#pragma unroll
        for (int mm = 0; mm < 2; ++mm)
#pragma unroll
            for (int nn = 0; nn < 4; ++nn)
                acc[mm][nn] = __builtin_amdgcn_wmma_f32_16x16x32_bf16(
                    false, fa[mm], false, fb[nn],
                    (short)0, acc[mm][nn], false, false);
    };

    // ---- prologue: stage tile 0 ----
    {
        uint4 a0, a1, b0, b1;
        loadGlobal(0, a0, a1, b0, b1);
        storeLds(0, a0, a1, b0, b1);
    }
    __syncthreads();

    constexpr int NK = KD / 32;
    int cur = 0;
    // branch-free steady state: always prefetch next tile
    for (int it = 0; it < NK - 1; ++it) {
        uint4 na0, na1, nb0, nb1;
        loadGlobal((it + 1) * 32, na0, na1, nb0, nb1);
        compute(cur);
        storeLds(cur ^ 1, na0, na1, nb0, nb1);
        __syncthreads();
        cur ^= 1;
    }
    // peeled last iteration: no prefetch, no barrier needed afterwards
    compute(cur);

    // ---- epilogue ----
#pragma unroll
    for (int mm = 0; mm < 2; ++mm) {
#pragma unroll
        for (int nn = 0; nn < 4; ++nn) {
            int colg = n0 + wn * 64 + nn * 16 + lr;
            float bv = bias[colg];
#pragma unroll
            for (int r = 0; r < 8; ++r) {
                int rowg = m0 + wm * 32 + mm * 16 + r + 8 * lh;
                float v = acc[mm][nn][r] + bv;
                if (EPI == 0) {
                    float g = 0.5f * v * (1.0f + erff(v * 0.70710678118654752f));
                    Hout[(size_t)rowg * ND + colg] = f2bf(g);
                } else {
                    float scale = probs[rowg * NEXP + expert];
                    float y = scale * v;
                    float* o = &Fout[(size_t)rowg * ND + colg];
                    if (initOut) *o = y;
                    else         *o = *o + y;
                }
            }
        }
    }
}

// ---------------------------------------------------------------------------
// Launch: convert x to bf16, transpose-convert w1/w2, router, then per-expert
// GEMM1 -> GEMM2.
// Workspace layout (≈113 MB):
//   probs f32 [8192,8]            256 KB
//   xbf   bf16 [8192,1024]         16 MB
//   w1t   bf16 [8,2048,1024]       32 MB   (w1[e]^T, K=C contiguous)
//   w2t   bf16 [8,1024,2048]       32 MB   (w2[e]^T, K=D contiguous)
//   hbf   bf16 [8192,2048]         32 MB   (reused across experts)
// ---------------------------------------------------------------------------
extern "C" void kernel_launch(void* const* d_in, const int* in_sizes, int n_in,
                              void* d_out, int out_size, void* d_ws, size_t ws_size,
                              hipStream_t stream) {
    const float* x        = (const float*)d_in[0];
    const float* w_router = (const float*)d_in[1];
    const float* b_router = (const float*)d_in[2];
    const float* w1       = (const float*)d_in[3];
    const float* b1       = (const float*)d_in[4];
    const float* w2       = (const float*)d_in[5];
    const float* b2       = (const float*)d_in[6];
    float* out = (float*)d_out;

    char* ws = (char*)d_ws;
    float*          probs = (float*)ws;                               // 262144 B
    unsigned short* xbf   = (unsigned short*)(ws + 262144);
    unsigned short* w1t   = xbf + (size_t)NTOK * CDIM;                // +16 MB
    unsigned short* w2t   = w1t + (size_t)NEXP * CDIM * DDIM;         // +32 MB
    unsigned short* hbf   = w2t + (size_t)NEXP * DDIM * CDIM;         // +32 MB

    const int nX = NTOK * CDIM;          // 8388608

    cvt_f32_bf16<<<(nX / 4 + 255) / 256, 256, 0, stream>>>(x, xbf, nX / 4);
    // w1 [E][C][D] -> w1t [E][D][C]
    cvt_transpose_bf16<<<dim3(DDIM / 64, CDIM / 64, NEXP), 256, 0, stream>>>(
        w1, w1t, CDIM, DDIM);
    // w2 [E][D][C] -> w2t [E][C][D]
    cvt_transpose_bf16<<<dim3(CDIM / 64, DDIM / 64, NEXP), 256, 0, stream>>>(
        w2, w2t, DDIM, CDIM);

    router_kernel<<<NTOK / 32, 256, 0, stream>>>(x, w_router, b_router, probs);

    for (int e = 0; e < NEXP; ++e) {
        // h = gelu(x @ w1[e] + b1[e])      [8192, 2048] bf16
        gemm_wmma<CDIM, DDIM, 0><<<dim3(DDIM / 128, NTOK / 128), 256, 0, stream>>>(
            xbf, w1t + (size_t)e * DDIM * CDIM, b1 + (size_t)e * DDIM,
            hbf, nullptr, nullptr, e, 0);
        // out (+)= probs[:,e] * (h @ w2[e] + b2[e])   [8192, 1024] f32
        gemm_wmma<DDIM, CDIM, 1><<<dim3(CDIM / 128, NTOK / 128), 256, 0, stream>>>(
            hbf, w2t + (size_t)e * CDIM * DDIM, b2 + (size_t)e * CDIM,
            nullptr, out, probs, e, (e == 0) ? 1 : 0);
    }
}
